// SAGENodeModel_39402029973520
// MI455X (gfx1250) — compile-verified, matrix operand order checked by reading, not run on hardware
//
#include <hip/hip_runtime.h>
#include <hip/hip_bf16.h>

typedef __attribute__((ext_vector_type(16))) _Float16 v16h;
typedef __attribute__((ext_vector_type(4)))  _Float16 v4h;
typedef __attribute__((ext_vector_type(8)))  float    v8f;

#define N_NODES_C 100000
#define N_EDGES_C 1600000
#define IN_DIM_C  128
#define HID_DIM_C 128
#define OUT_DIM_C 64
#define BN_EPS_C  1e-5f
#define LDA_PAD   136     // padded f16 row stride in LDS (16B aligned, spreads banks)
#define M_CHUNK   80      // rows per GEMM block (5 tiles of 16; 80 | 100000)

// ---------------------------------------------------------------------------
// zero fill
// ---------------------------------------------------------------------------
__global__ void zero_kernel(float* __restrict__ p, size_t n) {
  size_t i = (size_t)blockIdx.x * blockDim.x + threadIdx.x;
  size_t stride = (size_t)gridDim.x * blockDim.x;
  for (; i < n; i += stride) p[i] = 0.0f;
}

// ---------------------------------------------------------------------------
// one-time weight convert f32 -> f16
// ---------------------------------------------------------------------------
__global__ void cvt_f16_kernel(const float* __restrict__ s,
                               _Float16* __restrict__ d, int n) {
  int i = blockIdx.x * blockDim.x + threadIdx.x;
  if (i < n) d[i] = (_Float16)s[i];
}

// ---------------------------------------------------------------------------
// edge scatter (f32 features): agg[dst] += feat[src]; deg[dst] += 1
// 32 lanes per edge, float4/lane -> coalesced 512B gather, L2-resident
// ---------------------------------------------------------------------------
__global__ __launch_bounds__(256) void scatter_f32_kernel(
    const float* __restrict__ feat, const int* __restrict__ src,
    const int* __restrict__ dst, float* __restrict__ agg,
    float* __restrict__ deg, int nE) {
  int t = blockIdx.x * blockDim.x + threadIdx.x;
  int e = t >> 5;
  int lane = t & 31;
  if (e >= nE) return;
  int s = src[e];
  int d = dst[e];
  float4 v = ((const float4*)(feat + (size_t)s * 128))[lane];
  float* a = agg + (size_t)d * 128 + lane * 4;
  atomicAdd(a + 0, v.x);
  atomicAdd(a + 1, v.y);
  atomicAdd(a + 2, v.z);
  atomicAdd(a + 3, v.w);
  if (deg != nullptr && lane == 0) atomicAdd(deg + d, 1.0f);
}

// ---------------------------------------------------------------------------
// edge scatter (f16 features, f32 accumulate): halves gather bytes
// ---------------------------------------------------------------------------
__global__ __launch_bounds__(256) void scatter_f16_kernel(
    const _Float16* __restrict__ feat, const int* __restrict__ src,
    const int* __restrict__ dst, float* __restrict__ agg, int nE) {
  int t = blockIdx.x * blockDim.x + threadIdx.x;
  int e = t >> 5;
  int lane = t & 31;
  if (e >= nE) return;
  int s = src[e];
  int d = dst[e];
  v4h v = *(const v4h*)(feat + (size_t)s * 128 + lane * 4);
  float* a = agg + (size_t)d * 128 + lane * 4;
  atomicAdd(a + 0, (float)v[0]);
  atomicAdd(a + 1, (float)v[1]);
  atomicAdd(a + 2, (float)v[2]);
  atomicAdd(a + 3, (float)v[3]);
}

// ---------------------------------------------------------------------------
// WMMA fragment builders (ISA 7.12.2 16-bit A/B layout, wave32)
//   lane = (M or N) % 16 ; half = lane/16
//   frag[j]   <- K = kb + j        (j = 0..7)
//   frag[j+8] <- K = kb + 16 + j
//   where kb = ktile*32 + half*8
// ---------------------------------------------------------------------------
__device__ __forceinline__ v16h frag_from_lds(const _Float16* __restrict__ base,
                                              int row, int kb) {
  const _Float16* p = base + row * LDA_PAD + kb;
  v16h f;
#pragma unroll
  for (int j = 0; j < 8; ++j) f[j] = p[j];
#pragma unroll
  for (int j = 0; j < 8; ++j) f[8 + j] = p[16 + j];
  return f;
}

// B(k, o) = W[o, k]; weights pre-converted to f16 row-major [O][128]
__device__ __forceinline__ v16h frag_from_wf16(const _Float16* __restrict__ W,
                                               int o, int kb) {
  const _Float16* p = W + (size_t)o * 128 + kb;
  v16h f;
#pragma unroll
  for (int j = 0; j < 8; ++j) f[j] = p[j];
#pragma unroll
  for (int j = 0; j < 8; ++j) f[8 + j] = p[16 + j];
  return f;
}

// ---------------------------------------------------------------------------
// Layer 1: h = ReLU(BN( (agg/deg) @ W1l^T + b1l + x @ W1r^T )), h stored f16
// block = 80 nodes (5 tiles); 8 waves = 8 column tiles of HID_DIM=128;
// B fragments register-cached across the 5 M-tiles.
// ---------------------------------------------------------------------------
__global__ __launch_bounds__(256) void sage_gemm1_kernel(
    const float* __restrict__ agg, const float* __restrict__ deg,
    const float* __restrict__ x, const _Float16* __restrict__ W1l,
    const float* __restrict__ b1l, const _Float16* __restrict__ W1r,
    const float* __restrict__ gamma, const float* __restrict__ beta,
    const float* __restrict__ mean, const float* __restrict__ var,
    _Float16* __restrict__ h) {
  __shared__ _Float16 sA[M_CHUNK * LDA_PAD];
  __shared__ _Float16 sX[M_CHUNK * LDA_PAD];
  const int tid = threadIdx.x;
  const size_t nodebase = (size_t)blockIdx.x * M_CHUNK;

  // stage 80x128 of normalized-agg and x as f16 (8 contiguous elems/thread/iter)
#pragma unroll
  for (int it = 0; it < M_CHUNK / 16; ++it) {
    int base = tid * 8;            // 2048 elems = 16 rows per iteration
    int row = it * 16 + (base >> 7);
    int colb = base & 127;
    size_t g = (nodebase + row) * 128 + colb;
    float inv = 1.0f / fmaxf(deg[nodebase + row], 1.0f);
    _Float16* pa = &sA[row * LDA_PAD + colb];
    _Float16* px = &sX[row * LDA_PAD + colb];
#pragma unroll
    for (int j = 0; j < 8; ++j) pa[j] = (_Float16)(agg[g + j] * inv);
#pragma unroll
    for (int j = 0; j < 8; ++j) px[j] = (_Float16)(x[g + j]);
  }
  __syncthreads();

  const int wave = tid >> 5;
  const int lane = tid & 31;
  const int m = lane & 15;
  const int half = lane >> 4;
  const int col = wave * 16 + m;   // wave = column tile (HID_DIM/16 = 8)

  // register-cache B fragments for all 8 K-tiles (2 weight matrices x 4)
  v16h bl[4], br[4];
#pragma unroll
  for (int kt = 0; kt < 4; ++kt) {
    int kb = kt * 32 + half * 8;
    bl[kt] = frag_from_wf16(W1l, col, kb);
    br[kt] = frag_from_wf16(W1r, col, kb);
  }

  const float bb = b1l[col];
  const float mu = mean[col];
  const float sc = gamma[col] * rsqrtf(var[col] + BN_EPS_C);
  const float bt = beta[col];

  for (int mt = 0; mt < M_CHUNK / 16; ++mt) {
    const _Float16* sAb = sA + mt * 16 * LDA_PAD;
    const _Float16* sXb = sX + mt * 16 * LDA_PAD;
    v8f acc = {};
#pragma unroll
    for (int kt = 0; kt < 4; ++kt) {
      int kb = kt * 32 + half * 8;
      v16h a = frag_from_lds(sAb, m, kb);
      acc = __builtin_amdgcn_wmma_f32_16x16x32_f16(false, a, false, bl[kt],
                                                   (short)0, acc, false, false);
    }
#pragma unroll
    for (int kt = 0; kt < 4; ++kt) {
      int kb = kt * 32 + half * 8;
      v16h a = frag_from_lds(sXb, m, kb);
      acc = __builtin_amdgcn_wmma_f32_16x16x32_f16(false, a, false, br[kt],
                                                   (short)0, acc, false, false);
    }
#pragma unroll
    for (int r = 0; r < 8; ++r) {
      size_t row = nodebase + mt * 16 + r + half * 8;  // C/D: M = r + 8*half
      float v = acc[r] + bb;
      v = (v - mu) * sc + bt;
      v = fmaxf(v, 0.0f);
      h[row * 128 + col] = (_Float16)v;
    }
  }
}

// ---------------------------------------------------------------------------
// Layer 2: out = (agg/deg) @ W2l^T + b2l + h @ W2r^T   (OUT_DIM = 64, f32 out)
// block = 80 nodes; wave w -> column tile (w&3), M-tiles mt = (w>>2), step 2
// ---------------------------------------------------------------------------
__global__ __launch_bounds__(256) void sage_gemm2_kernel(
    const float* __restrict__ agg, const float* __restrict__ deg,
    const _Float16* __restrict__ hfeat, const _Float16* __restrict__ W2l,
    const float* __restrict__ b2l, const _Float16* __restrict__ W2r,
    float* __restrict__ out) {
  __shared__ _Float16 sA[M_CHUNK * LDA_PAD];
  __shared__ _Float16 sH[M_CHUNK * LDA_PAD];
  const int tid = threadIdx.x;
  const size_t nodebase = (size_t)blockIdx.x * M_CHUNK;

#pragma unroll
  for (int it = 0; it < M_CHUNK / 16; ++it) {
    int base = tid * 8;
    int row = it * 16 + (base >> 7);
    int colb = base & 127;
    size_t g = (nodebase + row) * 128 + colb;
    float inv = 1.0f / fmaxf(deg[nodebase + row], 1.0f);
    _Float16* pa = &sA[row * LDA_PAD + colb];
    _Float16* ph = &sH[row * LDA_PAD + colb];
#pragma unroll
    for (int j = 0; j < 8; ++j) pa[j] = (_Float16)(agg[g + j] * inv);
#pragma unroll
    for (int j = 0; j < 8; ++j) ph[j] = hfeat[g + j];   // f16 copy
  }
  __syncthreads();

  const int wave = tid >> 5;
  const int lane = tid & 31;
  const int m = lane & 15;
  const int half = lane >> 4;
  const int msub = wave >> 2;           // 0..1: even / odd M-tiles
  const int col = (wave & 3) * 16 + m;  // OUT_DIM/16 = 4 column tiles

  v16h bl[4], br[4];
#pragma unroll
  for (int kt = 0; kt < 4; ++kt) {
    int kb = kt * 32 + half * 8;
    bl[kt] = frag_from_wf16(W2l, col, kb);
    br[kt] = frag_from_wf16(W2r, col, kb);
  }
  const float bb = b2l[col];

  for (int mt = msub; mt < M_CHUNK / 16; mt += 2) {
    const _Float16* sAb = sA + mt * 16 * LDA_PAD;
    const _Float16* sHb = sH + mt * 16 * LDA_PAD;
    v8f acc = {};
#pragma unroll
    for (int kt = 0; kt < 4; ++kt) {
      int kb = kt * 32 + half * 8;
      v16h a = frag_from_lds(sAb, m, kb);
      acc = __builtin_amdgcn_wmma_f32_16x16x32_f16(false, a, false, bl[kt],
                                                   (short)0, acc, false, false);
    }
#pragma unroll
    for (int kt = 0; kt < 4; ++kt) {
      int kb = kt * 32 + half * 8;
      v16h a = frag_from_lds(sHb, m, kb);
      acc = __builtin_amdgcn_wmma_f32_16x16x32_f16(false, a, false, br[kt],
                                                   (short)0, acc, false, false);
    }
#pragma unroll
    for (int r = 0; r < 8; ++r) {
      size_t row = nodebase + mt * 16 + r + half * 8;
      out[row * 64 + col] = acc[r] + bb;
    }
  }
}

// ---------------------------------------------------------------------------
// launch
// ---------------------------------------------------------------------------
extern "C" void kernel_launch(void* const* d_in, const int* in_sizes, int n_in,
                              void* d_out, int out_size, void* d_ws, size_t ws_size,
                              hipStream_t stream) {
  const float* x     = (const float*)d_in[0];
  const int*   edge  = (const int*)d_in[1];   // [2, E]: row0 = src, row1 = dst
  const float* W1l   = (const float*)d_in[2];
  const float* b1l   = (const float*)d_in[3];
  const float* W1r   = (const float*)d_in[4];
  const float* gamma = (const float*)d_in[5];
  const float* beta  = (const float*)d_in[6];
  const float* mean  = (const float*)d_in[7];
  const float* var   = (const float*)d_in[8];
  const float* W2l   = (const float*)d_in[9];
  const float* b2l   = (const float*)d_in[10];
  const float* W2r   = (const float*)d_in[11];
  const int* src = edge;
  const int* dst = edge + N_EDGES_C;

  // workspace layout:
  //   agg  f32 [N*128]   51.2 MB
  //   deg  f32 [N]        0.4 MB
  //   h    f16 [N*128]   25.6 MB
  //   W1lh/W1rh f16 [128*128], W2lh/W2rh f16 [64*128]
  float*     agg  = (float*)d_ws;
  float*     deg  = agg + (size_t)N_NODES_C * 128;
  _Float16*  h    = (_Float16*)(deg + N_NODES_C);
  _Float16*  W1lh = h + (size_t)N_NODES_C * 128;
  _Float16*  W1rh = W1lh + 128 * 128;
  _Float16*  W2lh = W1rh + 128 * 128;
  _Float16*  W2rh = W2lh + 64 * 128;
  float*     out  = (float*)d_out;

  // one-time weight conversion to f16 (96 KB total, trivial)
  cvt_f16_kernel<<<(128 * 128 + 255) / 256, 256, 0, stream>>>(W1l, W1lh, 128 * 128);
  cvt_f16_kernel<<<(128 * 128 + 255) / 256, 256, 0, stream>>>(W1r, W1rh, 128 * 128);
  cvt_f16_kernel<<<(64 * 128 + 255) / 256, 256, 0, stream>>>(W2l, W2lh, 64 * 128);
  cvt_f16_kernel<<<(64 * 128 + 255) / 256, 256, 0, stream>>>(W2r, W2rh, 64 * 128);

  const int scatter_blocks = (int)(((long long)N_EDGES_C * 32 + 255) / 256);

  // layer 1
  zero_kernel<<<8192, 256, 0, stream>>>(agg, (size_t)N_NODES_C * 128 + N_NODES_C);
  scatter_f32_kernel<<<scatter_blocks, 256, 0, stream>>>(x, src, dst, agg, deg,
                                                         N_EDGES_C);
  sage_gemm1_kernel<<<N_NODES_C / M_CHUNK, 256, 0, stream>>>(
      agg, deg, x, W1lh, b1l, W1rh, gamma, beta, mean, var, h);

  // layer 2 (degree unchanged; re-zero only agg)
  zero_kernel<<<8192, 256, 0, stream>>>(agg, (size_t)N_NODES_C * 128);
  scatter_f16_kernel<<<scatter_blocks, 256, 0, stream>>>(h, src, dst, agg,
                                                         N_EDGES_C);
  sage_gemm2_kernel<<<N_NODES_C / M_CHUNK, 256, 0, stream>>>(agg, deg, h, W2lh,
                                                             b2l, W2rh, out);
}
